// Block_16174846836992
// MI455X (gfx1250) — matrix operand order repaired and checked by reading, not compiled, not run on hardware
//
#include <hip/hip_runtime.h>
#include <hip/hip_bf16.h>

// ---------------------------------------------------------------------------
// MaxViT-style recurrent block for MI455X (gfx1250, wave32, WMMA).
// GEMMs (qkv / attention / proj / MLP) run through v_wmma_f32_16x16x32_f16;
// GEMM A-tiles are staged global->LDS with CDNA5 async-DMA
// (global_load_async_to_lds_b128 / s_wait_asynccnt); convs + layernorms are
// bandwidth-bound kernels.
// ---------------------------------------------------------------------------

typedef __attribute__((ext_vector_type(16))) _Float16 v16h;
typedef __attribute__((ext_vector_type(8)))  float    v8f;

#define kB   2
#define kH   256
#define kW   256
#define kC   256
#define kTOK (kB * kH * kW)        // 131072 tokens
#define kP   8                     // window side
#define kL   64                    // tokens per window
#define kNH  8                     // heads
#define kDH  32                    // dim head
#define kNWIN (kTOK / kL)          // 2048 windows

__device__ __forceinline__ void wait_asynccnt0() {
#if __has_builtin(__builtin_amdgcn_s_wait_asynccnt)
  __builtin_amdgcn_s_wait_asynccnt(0);
#else
  asm volatile("s_wait_asynccnt 0x0" ::: "memory");
#endif
}

// window-major row for spatial token t
__device__ __forceinline__ int spatial_to_win(int t) {
  int b = t >> 16, y = (t >> 8) & 255, x = t & 255;
  int wy = y >> 3, py = y & 7, wx = x >> 3, px = x & 7;
  int win = b * 1024 + wy * 32 + wx;
  return win * kL + py * kP + px;
}
// spatial token for window-major row r
__device__ __forceinline__ int win_to_spatial(int r) {
  int win = r >> 6, l = r & 63;
  int b = win >> 10, rem = win & 1023;
  int wy = rem >> 5, wx = rem & 31;
  int y = wy * 8 + (l >> 3), x = wx * 8 + (l & 7);
  return (b << 16) + (y << 8) + x;
}

__device__ __forceinline__ float gelu_f(float x) {
  float x3 = x * x * x;
  return 0.5f * x * (1.f + tanhf(0.79788456080286535588f * (x + 0.044715f * x3)));
}

// ---------------------------------------------------------------------------
// depthwise 3x3 conv, NCHW input -> NHWC output, + relu
// ---------------------------------------------------------------------------
__global__ __launch_bounds__(256)
void dwconv3x3_nchw_relu(const float* __restrict__ x, const float* __restrict__ w,
                         const float* __restrict__ bi, float* __restrict__ out) {
  int t = blockIdx.x, c = threadIdx.x;
  int b = t >> 16, y = (t >> 8) & 255, xx = t & 255;
  const float* xp = x + ((size_t)(b * kC + c)) * (kH * kW);
  float s = bi[c];
  #pragma unroll
  for (int ky = 0; ky < 3; ++ky) {
    int yy = y + ky - 1;
    if (yy < 0 || yy >= kH) continue;
    #pragma unroll
    for (int kx = 0; kx < 3; ++kx) {
      int xc = xx + kx - 1;
      if (xc < 0 || xc >= kW) continue;
      s += xp[yy * kW + xc] * w[c * 9 + ky * 3 + kx];
    }
  }
  out[(size_t)t * kC + c] = fmaxf(s, 0.f);
}

// ---------------------------------------------------------------------------
// grouped 7x7 "combine" conv: groups=C, 2 input channels/group (x_conv, h)
// ---------------------------------------------------------------------------
__global__ __launch_bounds__(256)
void comb_conv7x7(const float* __restrict__ a, const float* __restrict__ h,
                  const float* __restrict__ w, const float* __restrict__ bi,
                  float* __restrict__ out) {
  int t = blockIdx.x, c = threadIdx.x;
  int b = t >> 16, y = (t >> 8) & 255, xx = t & 255;
  float s = bi[c];
  for (int ky = 0; ky < 7; ++ky) {
    int yy = y + ky - 3;
    if (yy < 0 || yy >= kH) continue;
    for (int kx = 0; kx < 7; ++kx) {
      int xc = xx + kx - 3;
      if (xc < 0 || xc >= kW) continue;
      size_t p = ((size_t)(b * (kH * kW) + yy * kW + xc)) * kC + c;
      s += a[p] * w[c * 98 + ky * 7 + kx] + h[p] * w[c * 98 + 49 + ky * 7 + kx];
    }
  }
  out[(size_t)t * kC + c] = s;   // relu+LN applied by ln kernel
}

// ---------------------------------------------------------------------------
// depthwise 7x7 conv, NHWC -> NHWC, + relu
// ---------------------------------------------------------------------------
__global__ __launch_bounds__(256)
void dwconv7x7_nhwc_relu(const float* __restrict__ in, const float* __restrict__ w,
                         const float* __restrict__ bi, float* __restrict__ out) {
  int t = blockIdx.x, c = threadIdx.x;
  int b = t >> 16, y = (t >> 8) & 255, xx = t & 255;
  float s = bi[c];
  for (int ky = 0; ky < 7; ++ky) {
    int yy = y + ky - 3;
    if (yy < 0 || yy >= kH) continue;
    for (int kx = 0; kx < 7; ++kx) {
      int xc = xx + kx - 3;
      if (xc < 0 || xc >= kW) continue;
      s += in[((size_t)(b * (kH * kW) + yy * kW + xc)) * kC + c] * w[c * 49 + ky * 7 + kx];
    }
  }
  out[(size_t)t * kC + c] = fmaxf(s, 0.f);
}

// ---------------------------------------------------------------------------
// grouped 3x3 final conv: 64 groups, 4 in-ch / 1 out-ch, NHWC in -> NCHW out
// ---------------------------------------------------------------------------
__global__ __launch_bounds__(64)
void convf3x3_relu(const float* __restrict__ in, const float* __restrict__ w,
                   const float* __restrict__ bi, float* __restrict__ out) {
  int t = blockIdx.x, oc = threadIdx.x;   // oc in [0,64)
  int b = t >> 16, y = (t >> 8) & 255, xx = t & 255;
  float s = bi[oc];
  #pragma unroll
  for (int ic = 0; ic < 4; ++ic) {
    int cin = oc * 4 + ic;
    for (int ky = 0; ky < 3; ++ky) {
      int yy = y + ky - 1;
      if (yy < 0 || yy >= kH) continue;
      for (int kx = 0; kx < 3; ++kx) {
        int xc = xx + kx - 1;
        if (xc < 0 || xc >= kW) continue;
        s += in[((size_t)(b * (kH * kW) + yy * kW + xc)) * kC + cin] *
             w[oc * 36 + ic * 9 + ky * 3 + kx];
      }
    }
  }
  out[((size_t)(b * 64 + oc)) * (kH * kW) + y * kW + xx] = fmaxf(s, 0.f);
}

// ---------------------------------------------------------------------------
// layernorm over C=256, block = one token, optional relu-in / f16-out / reorder
// ---------------------------------------------------------------------------
template <bool RELU, bool OUT16, bool REORDER>
__global__ __launch_bounds__(256)
void ln_kernel(const float* __restrict__ in, float* __restrict__ o32,
               _Float16* __restrict__ o16, const float* __restrict__ g,
               const float* __restrict__ bta) {
  __shared__ float red[256];
  int t = blockIdx.x, c = threadIdx.x;
  float x = in[(size_t)t * kC + c];
  if (RELU) x = fmaxf(x, 0.f);
  red[c] = x;
  __syncthreads();
  for (int s = 128; s > 0; s >>= 1) { if (c < s) red[c] += red[c + s]; __syncthreads(); }
  float mean = red[0] * (1.f / kC);
  __syncthreads();
  float d = x - mean;
  red[c] = d * d;
  __syncthreads();
  for (int s = 128; s > 0; s >>= 1) { if (c < s) red[c] += red[c + s]; __syncthreads(); }
  float var = red[0] * (1.f / kC);
  float y = d * rsqrtf(var + 1e-5f) * g[c] + bta[c];
  int row = REORDER ? spatial_to_win(t) : t;
  if (OUT16) o16[(size_t)row * kC + c] = (_Float16)y;
  else       o32[(size_t)row * kC + c] = y;
}

// ---------------------------------------------------------------------------
// WMMA GEMM:  out = A(MxK, f16) @ Bw(NxK, f32 weights)^T + bias, epilogues
//   EPI 0: bias                 -> f16 at gRow
//   EPI 1: bias + GELU          -> f16 at gRow
//   EPI 2: bias + residual, window->spatial remap -> f32
//   EPI 3: bias                 -> f32 at gRow
// Block 256 thr = 8 waves, 64x64 output tile, K staged in 32-chunks.
// A tile staged with CDNA5 async DMA (global_load_async_to_lds_b128),
// B tile staged via VALU (f32 -> f16 conversion in flight).
// ---------------------------------------------------------------------------
#define EP_BIAS_F16   0
#define EP_GELU_F16   1
#define EP_RES_F32MAP 2
#define EP_BIAS_F32   3

template <int EPI>
__global__ __launch_bounds__(256)
void wmma_gemm(const _Float16* __restrict__ Ap, const float* __restrict__ Bw,
               const float* __restrict__ bias, const float* __restrict__ resid,
               float* __restrict__ o32, _Float16* __restrict__ o16,
               int M, int N, int K) {
  __shared__ _Float16 As[64][40];   // [m][k], padded (row stride 80B, 16B aligned)
  __shared__ _Float16 Bs[64][40];   // [n][k] (== B^T), padded
  int tid  = threadIdx.x;
  int wave = tid >> 5, lane = tid & 31;
  int m0 = blockIdx.x * 64, n0 = blockIdx.y * 64;
  int mi  = wave & 3;             // M sub-tile of this wave
  int nig = (wave >> 2) * 2;      // first of two N sub-tiles

  v8f acc0 = {0.f, 0.f, 0.f, 0.f, 0.f, 0.f, 0.f, 0.f};
  v8f acc1 = {0.f, 0.f, 0.f, 0.f, 0.f, 0.f, 0.f, 0.f};

  int Ml = lane & 15;
  int abase = (lane >> 4) * 16;   // A half-K select
  int kb    = (lane >> 4) * 16;   // B half-K select

  // per-thread async-DMA slot for the A tile: 16B each, 256 thr == 4096B tile
  int ar = tid >> 2;              // A row
  int ac = (tid & 3) * 8;         // A col (f16 units)
  unsigned lds_a = (unsigned)(uintptr_t)&As[ar][ac];

  for (int k0 = 0; k0 < K; k0 += 32) {
    // ---- A tile: global -> LDS async DMA (ASYNCcnt-tracked, no VGPR data) --
    {
      unsigned long long ga =
          (unsigned long long)(uintptr_t)(Ap + (size_t)(m0 + ar) * K + k0 + ac);
      asm volatile("global_load_async_to_lds_b128 %0, %1, off"
                   :: "v"(lds_a), "v"(ga) : "memory");
    }
    // ---- B tile: stage + f32->f16 convert via VALU ----
    for (int i = tid; i < 64 * 32; i += 256) {
      int r = i >> 5, cc = i & 31;
      Bs[r][cc] = (_Float16)Bw[(size_t)(n0 + r) * K + k0 + cc];
    }
    if (k0 + 32 < K) {  // gfx1250 global_prefetch_b8 of next weight tile
      __builtin_prefetch((const char*)(Bw + (size_t)(n0 + (tid >> 2)) * K + k0 + 32), 0, 1);
    }
    wait_asynccnt0();   // this wave's DMA landed in LDS
    __syncthreads();    // cross-wave visibility of As/Bs

    v16h a;
    #pragma unroll
    for (int p = 0; p < 8; ++p) {
      a[p]     = As[mi * 16 + Ml][abase + p];
      a[p + 8] = As[mi * 16 + Ml][abase + 8 + p];
    }
    v16h b0, b1;
    #pragma unroll
    for (int i = 0; i < 16; ++i) {
      b0[i] = Bs[nig * 16 + Ml][kb + i];
      b1[i] = Bs[(nig + 1) * 16 + Ml][kb + i];
    }
    acc0 = __builtin_amdgcn_wmma_f32_16x16x32_f16(false, a, false, b0, (short)0, acc0, false, false);
    acc1 = __builtin_amdgcn_wmma_f32_16x16x32_f16(false, a, false, b1, (short)0, acc1, false, false);
    __syncthreads();    // readers done before next iteration's DMA overwrites
  }

  int rowAdd = (lane >> 4) * 8;
  #pragma unroll
  for (int tsub = 0; tsub < 2; ++tsub) {
    v8f acc = tsub ? acc1 : acc0;
    int gcol = n0 + (nig + tsub) * 16 + Ml;
    float bv = bias[gcol];
    #pragma unroll
    for (int r = 0; r < 8; ++r) {
      int gRow = m0 + mi * 16 + r + rowAdd;
      float v = acc[r] + bv;
      if (EPI == EP_GELU_F16) {
        o16[(size_t)gRow * N + gcol] = (_Float16)gelu_f(v);
      } else if (EPI == EP_BIAS_F16) {
        o16[(size_t)gRow * N + gcol] = (_Float16)v;
      } else if (EPI == EP_RES_F32MAP) {
        int sr = win_to_spatial(gRow);
        v += resid[(size_t)sr * N + gcol];
        o32[(size_t)sr * N + gcol] = v;
      } else {
        o32[(size_t)gRow * N + gcol] = v;
      }
    }
  }
}

// ---------------------------------------------------------------------------
// windowed attention: block = 64 thr (2 waves = 2 heads), grid (2048, 4).
// qkv rows window-major, 768 cols: head h -> [q|k|v] at cols h*96 + {0,32,64}.
// ---------------------------------------------------------------------------
__global__ __launch_bounds__(64)
void window_attn_kernel(const _Float16* __restrict__ qkv, _Float16* __restrict__ out) {
  __shared__ float S[2][kL][kL];   // per-wave score / prob scratch (32 KB)
  int tid = threadIdx.x, wave = tid >> 5, lane = tid & 31;
  int win = blockIdx.x;
  int head = blockIdx.y * 2 + wave;
  const size_t base = (size_t)win * kL * 768;
  const int hq = head * 96, hk = hq + 32, hv = hq + 64;
  const int Ml = lane & 15;
  const int half8 = (lane >> 4) * 8;
  const int kb = (lane >> 4) * 16;

  // ---- S = Q @ K^T ----
  v16h aq[4];
  #pragma unroll
  for (int mi2 = 0; mi2 < 4; ++mi2) {
    size_t rr = base + (size_t)(mi2 * 16 + Ml) * 768 + hq;
    #pragma unroll
    for (int p = 0; p < 8; ++p) {
      aq[mi2][p]     = qkv[rr + kb + p];
      aq[mi2][p + 8] = qkv[rr + kb + 8 + p];
    }
  }
  v16h bk[4];
  #pragma unroll
  for (int ni = 0; ni < 4; ++ni) {
    size_t rr = base + (size_t)(ni * 16 + Ml) * 768 + hk;
    #pragma unroll
    for (int i = 0; i < 16; ++i) bk[ni][i] = qkv[rr + kb + i];
  }
  #pragma unroll
  for (int mi2 = 0; mi2 < 4; ++mi2) {
    #pragma unroll
    for (int ni = 0; ni < 4; ++ni) {
      v8f acc = {0.f, 0.f, 0.f, 0.f, 0.f, 0.f, 0.f, 0.f};
      acc = __builtin_amdgcn_wmma_f32_16x16x32_f16(false, aq[mi2], false, bk[ni],
                                                   (short)0, acc, false, false);
      #pragma unroll
      for (int r = 0; r < 8; ++r)
        S[wave][mi2 * 16 + r + ((lane >> 4) * 8)][ni * 16 + Ml] = acc[r];
    }
  }
  __syncthreads();

  // ---- softmax (scale 1/sqrt(32)), 2 rows per lane ----
  const float scale = 0.17677669529663687f;
  #pragma unroll
  for (int rr = 0; rr < 2; ++rr) {
    int row = lane * 2 + rr;
    float m = -1e30f;
    for (int j = 0; j < kL; ++j) m = fmaxf(m, S[wave][row][j] * scale);
    float s = 0.f;
    for (int j = 0; j < kL; ++j) {
      float e = __expf(S[wave][row][j] * scale - m);
      S[wave][row][j] = e;
      s += e;
    }
    float inv = 1.f / s;
    for (int j = 0; j < kL; ++j) S[wave][row][j] *= inv;
  }
  __syncthreads();

  // ---- O = P @ V ----
  #pragma unroll
  for (int mi2 = 0; mi2 < 4; ++mi2) {
    #pragma unroll
    for (int ni = 0; ni < 2; ++ni) {
      v8f acc = {0.f, 0.f, 0.f, 0.f, 0.f, 0.f, 0.f, 0.f};
      #pragma unroll
      for (int kc = 0; kc < 2; ++kc) {
        v16h ap;
        #pragma unroll
        for (int p = 0; p < 8; ++p) {
          ap[p]     = (_Float16)S[wave][mi2 * 16 + Ml][kc * 32 + kb + p];
          ap[p + 8] = (_Float16)S[wave][mi2 * 16 + Ml][kc * 32 + kb + 8 + p];
        }
        v16h bv;
        #pragma unroll
        for (int i = 0; i < 16; ++i)
          bv[i] = qkv[base + (size_t)(kc * 32 + kb + i) * 768 + hv + ni * 16 + Ml];
        acc = __builtin_amdgcn_wmma_f32_16x16x32_f16(false, ap, false, bv,
                                                     (short)0, acc, false, false);
      }
      #pragma unroll
      for (int r = 0; r < 8; ++r) {
        int row = mi2 * 16 + r + ((lane >> 4) * 8);
        out[((size_t)win * kL + row) * kC + head * kDH + ni * 16 + Ml] = (_Float16)acc[r];
      }
    }
  }
}

// ---------------------------------------------------------------------------
extern "C" void kernel_launch(void* const* d_in, const int* in_sizes, int n_in,
                              void* d_out, int out_size, void* d_ws, size_t ws_size,
                              hipStream_t stream) {
  const float* x       = (const float*)d_in[0];
  const float* h       = (const float*)d_in[1];
  const float* conv_w  = (const float*)d_in[2];
  const float* conv_b  = (const float*)d_in[3];
  const float* comb_w  = (const float*)d_in[4];
  const float* comb_b  = (const float*)d_in[5];
  const float* ln1_g   = (const float*)d_in[6];
  const float* ln1_b   = (const float*)d_in[7];
  const float* ln2_g   = (const float*)d_in[8];
  const float* ln2_b   = (const float*)d_in[9];
  const float* ln3_g   = (const float*)d_in[10];
  const float* ln3_b   = (const float*)d_in[11];
  const float* fsa_g   = (const float*)d_in[12];
  const float* fsa_b   = (const float*)d_in[13];
  const float* qkv_w   = (const float*)d_in[14];
  const float* qkv_b   = (const float*)d_in[15];
  const float* proj_w  = (const float*)d_in[16];
  const float* proj_b  = (const float*)d_in[17];
  const float* mlp_w1  = (const float*)d_in[18];
  const float* mlp_b1  = (const float*)d_in[19];
  const float* mlp_w2  = (const float*)d_in[20];
  const float* mlp_b2  = (const float*)d_in[21];
  const float* convg_w = (const float*)d_in[22];
  const float* convg_b = (const float*)d_in[23];
  const float* convh_w = (const float*)d_in[24];
  const float* convh_b = (const float*)d_in[25];
  const float* convf_w = (const float*)d_in[26];
  const float* convf_b = (const float*)d_in[27];

  // workspace layout (bytes)
  char* ws = (char*)d_ws;
  float*     Abuf = (float*)(ws);                       // 134,217,728  x_conv -> x_bsa -> convg out
  float*     Bbuf = (float*)(ws + 134217728ull);        // 134,217,728  xc -> mlp_out
  _Float16*  Fbuf = (_Float16*)(ws + 268435456ull);     //  67,108,864  LN f16 (window, then spatial)
  _Float16*  Qbuf = (_Float16*)(ws + 335544320ull);     // 268,435,456  qkv f16 -> mlp hidden f16
  _Float16*  Obuf = (_Float16*)(ws + 603979776ull);     //  67,108,864  attention out f16
  float*     out_x = (float*)d_out;                     // (2,64,256,256) NCHW
  float*     out_h = (float*)d_out + 8388608ull;        // (2,256,256,256) NHWC

  // 1) depthwise 3x3 + relu (NCHW -> NHWC), then LN1 in place
  dwconv3x3_nchw_relu<<<kTOK, 256, 0, stream>>>(x, conv_w, conv_b, Abuf);
  ln_kernel<false, false, false><<<kTOK, 256, 0, stream>>>(Abuf, Abuf, nullptr, ln1_g, ln1_b);

  // 2) grouped 7x7 combine with h, then relu + LN2 in place
  comb_conv7x7<<<kTOK, 256, 0, stream>>>(Abuf, h, comb_w, comb_b, Bbuf);
  ln_kernel<true, false, false><<<kTOK, 256, 0, stream>>>(Bbuf, Bbuf, nullptr, ln2_g, ln2_b);

  // 3) fsa LN -> f16, window-major layout
  ln_kernel<false, true, true><<<kTOK, 256, 0, stream>>>(Bbuf, nullptr, Fbuf, fsa_g, fsa_b);

  // 4) qkv GEMM (131072 x 256) @ (256 x 768)
  wmma_gemm<EP_BIAS_F16><<<dim3(kTOK / 64, 768 / 64), 256, 0, stream>>>(
      Fbuf, qkv_w, qkv_b, nullptr, nullptr, Qbuf, kTOK, 768, kC);

  // 5) windowed attention
  window_attn_kernel<<<dim3(kNWIN, 4), 64, 0, stream>>>(Qbuf, Obuf);

  // 6) proj GEMM + residual(xc) with window->spatial remap -> f32 x_bsa
  wmma_gemm<EP_RES_F32MAP><<<dim3(kTOK / 64, kC / 64), 256, 0, stream>>>(
      Obuf, proj_w, proj_b, Bbuf, Abuf, nullptr, kTOK, kC, kC);

  // 7) relu + LN3 -> f16 (spatial order)
  ln_kernel<true, true, false><<<kTOK, 256, 0, stream>>>(Abuf, nullptr, Fbuf, ln3_g, ln3_b);

  // 8) MLP: fc1 + GELU -> f16 hidden, fc2 -> f32
  wmma_gemm<EP_GELU_F16><<<dim3(kTOK / 64, 1024 / 64), 256, 0, stream>>>(
      Fbuf, mlp_w1, mlp_b1, nullptr, nullptr, Qbuf, kTOK, 1024, kC);
  wmma_gemm<EP_BIAS_F32><<<dim3(kTOK / 64, kC / 64), 256, 0, stream>>>(
      Qbuf, mlp_w2, mlp_b2, nullptr, Bbuf, nullptr, kTOK, kC, 1024);

  // 9) output heads
  dwconv7x7_nhwc_relu<<<kTOK, 256, 0, stream>>>(Bbuf, convg_w, convg_b, Abuf);
  dwconv7x7_nhwc_relu<<<kTOK, 256, 0, stream>>>(Abuf, convh_w, convh_b, out_h);
  convf3x3_relu<<<kTOK, 64, 0, stream>>>(Abuf, convf_w, convf_b, out_x);
}
